// pointnet2_reduce_67027259621498
// MI455X (gfx1250) — matrix operand order repaired and checked by reading, not compiled, not run on hardware
//
#include <hip/hip_runtime.h>

typedef __attribute__((ext_vector_type(16))) _Float16 v16h;
typedef __attribute__((ext_vector_type(8)))  _Float16 h8;
typedef __attribute__((ext_vector_type(8)))  float    v8f;

#define EPSBN 1e-5f

struct LayerP { const float *w, *b, *g, *bta, *m, *v; };     // raw params (prep input)
struct ConvL  { const _Float16* w; const float2* st; };      // f16 weights + folded BN

// ---------------------------------------------------------------------------
// WMMA fragment layouts (wave32, v_wmma_f32_16x16x32_f16):
//  A 16x32 f16: lane L<16 -> M=L; VGPRs 0-3 hold K=grp*8..grp*8+7, VGPRs 4-7 K=+16.
//               => two contiguous 8-half runs per lane -> 2x ds_load_b128.
//  B 32x16 f16: lane -> N=L%16; K = grp*8..grp*8+7 and +16 (same as A) -> 2x global b128.
//  C/D f32 16x16: lane L -> N=L%16, VGPR r -> M = r + 8*(L/16).
// ---------------------------------------------------------------------------
__device__ __forceinline__ v8f mlp_tile(const _Float16* lds_in, int strideIn,
                                        int kChunks, int cinPad,
                                        const _Float16* Wh, int m0, int n0, unsigned lane) {
  v8f acc = {0.f,0.f,0.f,0.f,0.f,0.f,0.f,0.f};
  const int l16 = (int)(lane & 15u);
  const int grp = (int)(lane >> 4);
  const _Float16* arow = lds_in + (m0 + l16) * strideIn;
  const _Float16* wrow = Wh + (size_t)(n0 + l16) * (size_t)cinPad;
  for (int kc = 0; kc < kChunks; ++kc) {
    const int kb = kc * 32;
    const h8 a0 = *(const h8*)(arow + kb + grp * 8);
    const h8 a1 = *(const h8*)(arow + kb + 16 + grp * 8);
    const h8 b0 = *(const h8*)(wrow + kb + grp * 8);
    const h8 b1 = *(const h8*)(wrow + kb + 16 + grp * 8);
    v16h a, b;
#pragma unroll
    for (int i = 0; i < 8; ++i) {
      a[i] = a0[i]; a[i + 8] = a1[i];
      b[i] = b0[i]; b[i + 8] = b1[i];
    }
    acc = __builtin_amdgcn_wmma_f32_16x16x32_f16(false, a, false, b, (short)0, acc, false, false);
  }
  return acc;
}

// folded bias+batchnorm (precomputed scale/shift) + relu
__device__ __forceinline__ void epilogue(v8f& acc, const float2* st, int n) {
  const float2 sv = st[n];
#pragma unroll
  for (int i = 0; i < 8; ++i) {
    float y = acc[i] * sv.x + sv.y;
    acc[i] = y > 0.f ? y : 0.f;
  }
}

__device__ __forceinline__ void store_tile(_Float16* lds_out, int strideOut,
                                           int m0, int n0, unsigned lane, const v8f& acc) {
  const int n   = n0 + (int)(lane & 15u);
  const int mgp = (int)(lane >> 4) * 8;
#pragma unroll
  for (int i = 0; i < 8; ++i)
    lds_out[(m0 + i + mgp) * strideOut + n] = (_Float16)acc[i];
}

__device__ __forceinline__ void mlp_layer(const _Float16* in, int sIn, _Float16* out, int sOut,
                                          int rows, int cinPad, int cout,
                                          ConvL L, unsigned lane, int wave, int nwaves) {
  const int MT = rows >> 4, NT = cout >> 4, kChunks = cinPad >> 5;
  for (int t = wave; t < MT * NT; t += nwaves) {
    const int mt = t / NT, nt = t % NT;
    v8f acc = mlp_tile(in, sIn, kChunks, cinPad, L.w, mt * 16, nt * 16, lane);
    epilogue(acc, L.st, nt * 16 + (int)(lane & 15u));
    store_tile(out, sOut, mt * 16, nt * 16, lane, acc);
  }
}

// Final layer fused with max-pool over all rows; one wave owns whole column tiles.
__device__ __forceinline__ void mlp_final_max(const _Float16* in, int sIn,
                                              int rows, int cinPad, int cout,
                                              ConvL L, float* outBase,
                                              unsigned lane, int wave, int nwaves) {
  const int MT = rows >> 4, NT = cout >> 4, kChunks = cinPad >> 5;
  for (int nt = wave; nt < NT; nt += nwaves) {
    float colmax = 0.f;  // relu output >= 0
    for (int mt = 0; mt < MT; ++mt) {
      v8f acc = mlp_tile(in, sIn, kChunks, cinPad, L.w, mt * 16, nt * 16, lane);
      epilogue(acc, L.st, nt * 16 + (int)(lane & 15u));
      float lm = acc[0];
#pragma unroll
      for (int i = 1; i < 8; ++i) lm = fmaxf(lm, acc[i]);
      lm = fmaxf(lm, __shfl_xor(lm, 16, 32));  // lanes L and L+16 share column
      colmax = fmaxf(colmax, lm);
    }
    if (lane < 16u) outBase[nt * 16 + (int)lane] = colmax;
  }
}

// ---------------------------------------------------------------------------
// Prep: convert one conv layer's weights to f16 (zero-padded cin) and fold BN
// ---------------------------------------------------------------------------
__global__ void __launch_bounds__(256)
prep_layer_kernel(const float* __restrict__ w, const float* __restrict__ b,
                  const float* __restrict__ g, const float* __restrict__ bta,
                  const float* __restrict__ m, const float* __restrict__ v,
                  _Float16* __restrict__ wh, float2* __restrict__ st,
                  int cout, int cin, int cinPad) {
  const int total = cout * cinPad;
  for (int j = blockIdx.x * 256 + (int)threadIdx.x; j < total; j += gridDim.x * 256) {
    const int n = j / cinPad, k = j % cinPad;
    wh[j] = (k < cin) ? (_Float16)w[(size_t)n * cin + k] : (_Float16)0.f;
  }
  const int i = blockIdx.x * 256 + (int)threadIdx.x;
  if (i < cout) {
    const float s = g[i] * rsqrtf(v[i] + EPSBN);
    st[i] = make_float2(s, (b[i] - m[i]) * s + bta[i]);
  }
}

// ---------------------------------------------------------------------------
// Kernel 1: farthest point sampling, one workgroup per batch (sequential scan)
// ---------------------------------------------------------------------------
__global__ void __launch_bounds__(512)
fps_kernel(const float* __restrict__ xyz, float* __restrict__ new_xyz, float* __restrict__ feat) {
  __shared__ float xl[4096], yl[4096], zl[4096];
  __shared__ float rv[16];
  __shared__ int   ri[16];
  __shared__ int   sfar;
  const int b = blockIdx.x;
  const int tid = threadIdx.x;
  const float* xb = xyz + (size_t)b * 3 * 4096;
  for (int n = tid; n < 4096; n += 512) {
    xl[n] = xb[n]; yl[n] = xb[4096 + n]; zl[n] = xb[8192 + n];
  }
  // zero SA3 concat pad channels [323,352) for this batch
  for (int i = tid; i < 512 * 29; i += 512) {
    int r = i / 29, c = 323 + i % 29;
    feat[((size_t)b * 512 + r) * 352 + c] = 0.f;
  }
  float dist[8];
#pragma unroll
  for (int j = 0; j < 8; ++j) dist[j] = 1e10f;
  if (tid == 0) sfar = 0;
  __syncthreads();
  for (int it = 0; it < 512; ++it) {
    const int far = sfar;
    const float cx = xl[far], cy = yl[far], cz = zl[far];
    if (tid == 0) {
      float* nx = new_xyz + ((size_t)b * 512 + it) * 3;
      nx[0] = cx; nx[1] = cy; nx[2] = cz;
      float* fr = feat + ((size_t)b * 512 + it) * 352;
      fr[0] = cx; fr[1] = cy; fr[2] = cz;
    }
    float bestv = -1.f; int besti = 0;
#pragma unroll
    for (int j = 0; j < 8; ++j) {
      const int n = tid + j * 512;
      const float dx = xl[n] - cx, dy = yl[n] - cy, dz = zl[n] - cz;
      const float d = dx * dx + dy * dy + dz * dz;
      dist[j] = fminf(dist[j], d);
      if (dist[j] > bestv) { bestv = dist[j]; besti = n; }
    }
    for (int off = 16; off > 0; off >>= 1) {
      const float ov = __shfl_xor(bestv, off, 32);
      const int   oi = __shfl_xor(besti, off, 32);
      if (ov > bestv || (ov == bestv && oi < besti)) { bestv = ov; besti = oi; }
    }
    if ((tid & 31) == 0) { rv[tid >> 5] = bestv; ri[tid >> 5] = besti; }
    __syncthreads();
    if (tid < 16) {
      bestv = rv[tid]; besti = ri[tid];
      for (int off = 8; off > 0; off >>= 1) {
        const float ov = __shfl_xor(bestv, off, 32);
        const int   oi = __shfl_xor(besti, off, 32);
        if (ov > bestv || (ov == bestv && oi < besti)) { bestv = ov; besti = oi; }
      }
      if (tid == 0) sfar = besti;
    }
    __syncthreads();
  }
}

// ---------------------------------------------------------------------------
// Kernel 2: ball query for all three radii, one wave per (b,q)
// ---------------------------------------------------------------------------
__global__ void __launch_bounds__(256)
ballquery_kernel(const float* __restrict__ xyz, const float* __restrict__ new_xyz,
                 float* __restrict__ g1, float* __restrict__ g2, float* __restrict__ g3) {
  const int gq = blockIdx.x * 8 + (int)(threadIdx.x >> 5);
  const unsigned lane = threadIdx.x & 31u;
  const int b = gq >> 9;
  const float* xb = xyz + (size_t)b * 3 * 4096;
  const float* nx = new_xyz + (size_t)gq * 3;
  const float cx = nx[0], cy = nx[1], cz = nx[2];
  float* o1 = g1 + (size_t)gq * 16 * 3;
  float* o2 = g2 + (size_t)gq * 32 * 3;
  float* o3 = g3 + (size_t)gq * 128 * 3;
  int base1 = 0, base2 = 0, base3 = 0;
  float f1x = 0, f1y = 0, f1z = 0, f2x = 0, f2y = 0, f2z = 0, f3x = 0, f3y = 0, f3z = 0;
  const float r1s = 0.01f, r2s = 0.04f, r3s = 0.16f;
  const unsigned below = (1u << lane) - 1u;
  for (int cb = 0; cb < 4096; cb += 32) {
    const int n = cb + (int)lane;
    const float dx = xb[n] - cx, dy = xb[4096 + n] - cy, dz = xb[8192 + n] - cz;
    const float d = dx * dx + dy * dy + dz * dz;
    const unsigned m1 = (unsigned)__ballot(d <= r1s);
    const unsigned m2 = (unsigned)__ballot(d <= r2s);
    const unsigned m3 = (unsigned)__ballot(d <= r3s);
    if (base1 < 16 && m1) {
      if (d <= r1s) {
        const int slot = base1 + __popc(m1 & below);
        if (slot < 16) { float* o = o1 + slot * 3; o[0] = dx; o[1] = dy; o[2] = dz; }
      }
      if (base1 == 0) {
        const int fl = __ffs(m1) - 1;
        f1x = __shfl(dx, fl, 32); f1y = __shfl(dy, fl, 32); f1z = __shfl(dz, fl, 32);
      }
      base1 += __popc(m1);
    }
    if (base2 < 32 && m2) {
      if (d <= r2s) {
        const int slot = base2 + __popc(m2 & below);
        if (slot < 32) { float* o = o2 + slot * 3; o[0] = dx; o[1] = dy; o[2] = dz; }
      }
      if (base2 == 0) {
        const int fl = __ffs(m2) - 1;
        f2x = __shfl(dx, fl, 32); f2y = __shfl(dy, fl, 32); f2z = __shfl(dz, fl, 32);
      }
      base2 += __popc(m2);
    }
    if (base3 < 128 && m3) {
      if (d <= r3s) {
        const int slot = base3 + __popc(m3 & below);
        if (slot < 128) { float* o = o3 + slot * 3; o[0] = dx; o[1] = dy; o[2] = dz; }
      }
      if (base3 == 0) {
        const int fl = __ffs(m3) - 1;
        f3x = __shfl(dx, fl, 32); f3y = __shfl(dy, fl, 32); f3z = __shfl(dz, fl, 32);
      }
      base3 += __popc(m3);
    }
    if (base1 >= 16 && base2 >= 32 && base3 >= 128) break;
  }
  for (int s = base1 + (int)lane; s < 16; s += 32)  { float* o = o1 + s * 3; o[0] = f1x; o[1] = f1y; o[2] = f1z; }
  for (int s = base2 + (int)lane; s < 32; s += 32)  { float* o = o2 + s * 3; o[0] = f2x; o[1] = f2y; o[2] = f2z; }
  for (int s = base3 + (int)lane; s < 128; s += 32) { float* o = o3 + s * 3; o[0] = f3x; o[1] = f3y; o[2] = f3z; }
}

// ---------------------------------------------------------------------------
// Kernel 3 (x3 branches): grouped 3-layer MLP + max-pool per group (WMMA)
// ---------------------------------------------------------------------------
template<int KPTS, int C1, int C2, int C3, int S0, int S1, int NWAVES>
__global__ void __launch_bounds__(NWAVES * 32)
sa1_branch_kernel(const float* __restrict__ grouped, ConvL L0, ConvL L1, ConvL L2,
                  float* __restrict__ feat, int chanOff) {
  __shared__ _Float16 buf0[KPTS * S0];
  __shared__ _Float16 buf1[KPTS * S1];
  const int gq = blockIdx.x;  // b*512+q
  const unsigned tid = threadIdx.x, lane = tid & 31u;
  const int wave = (int)(tid >> 5);
  if (tid == 0) {
    __builtin_prefetch(L0.w, 0, 0);
    __builtin_prefetch(L1.w, 0, 0);
    __builtin_prefetch(L2.w, 0, 0);
  }
  const float* gin = grouped + (size_t)gq * KPTS * 3;
  for (int i = (int)tid; i < KPTS * 32; i += NWAVES * 32) {
    const int r = i >> 5, c = i & 31;
    buf0[r * S0 + c] = (c < 3) ? (_Float16)gin[r * 3 + c] : (_Float16)0.f;
  }
  __syncthreads();
  mlp_layer(buf0, S0, buf1, S1, KPTS, 32, C1, L0, lane, wave, NWAVES);
  __syncthreads();
  mlp_layer(buf1, S1, buf0, S0, KPTS, C1, C2, L1, lane, wave, NWAVES);
  __syncthreads();
  mlp_final_max(buf0, S0, KPTS, C2, C3, L2, feat + (size_t)gq * 352 + chanOff, lane, wave, NWAVES);
}

// ---------------------------------------------------------------------------
// Kernel 4: SA3 global MLP (323->128->256->512) on 32-row chunks + partial max
// ---------------------------------------------------------------------------
__global__ void __launch_bounds__(256)
sa3_kernel(const float* __restrict__ feat, ConvL L0, ConvL L1, ConvL L2,
           float* __restrict__ part) {
  __shared__ _Float16 buf0[32 * 360];
  __shared__ _Float16 buf1[32 * 520];
  const int blk = blockIdx.x;                 // b*16 + chunk
  const int b = blk >> 4, chunk = blk & 15;
  const unsigned tid = threadIdx.x, lane = tid & 31u;
  const int wave = (int)(tid >> 5);
  if (tid == 0) {
    __builtin_prefetch(L0.w, 0, 0);
    __builtin_prefetch(L1.w, 0, 0);
    __builtin_prefetch(L2.w, 0, 0);
  }
  const float* fin = feat + ((size_t)(b * 512 + chunk * 32)) * 352;
  for (int i = (int)tid; i < 32 * 352; i += 256) {
    const int r = i / 352, c = i % 352;
    buf0[r * 360 + c] = (_Float16)fin[r * 352 + c];
  }
  __syncthreads();
  mlp_layer(buf0, 360, buf1, 520, 32, 352, 128, L0, lane, wave, 8);
  __syncthreads();
  mlp_layer(buf1, 520, buf0, 360, 32, 128, 256, L1, lane, wave, 8);
  __syncthreads();
  mlp_final_max(buf0, 360, 32, 256, 512, L2, part + (size_t)blk * 512, lane, wave, 8);
}

__global__ void __launch_bounds__(256)
sa3_reduce_kernel(const float* __restrict__ part, float* __restrict__ sa3max) {
  const int i = blockIdx.x * 256 + (int)threadIdx.x;  // 4096 = b*512+ch
  const int b = i >> 9, ch = i & 511;
  float mx = part[(size_t)(b * 16) * 512 + ch];
  for (int c = 1; c < 16; ++c) mx = fmaxf(mx, part[(size_t)(b * 16 + c) * 512 + ch]);
  sa3max[i] = mx;
}

// ---------------------------------------------------------------------------
// Kernel 5: fc2(+bn+relu) and fc3 (M=8 rows -> VALU)
// ---------------------------------------------------------------------------
__global__ void __launch_bounds__(512)
fc_kernel(const float* __restrict__ sa3max,
          const float* __restrict__ w2, const float* __restrict__ b2,
          const float* __restrict__ g2, const float* __restrict__ bt2,
          const float* __restrict__ m2, const float* __restrict__ v2,
          const float* __restrict__ w3, const float* __restrict__ b3,
          float* __restrict__ out) {
  __shared__ float x2[8 * 256];
  const int tid = (int)threadIdx.x;
  for (int o = tid; o < 2048; o += 512) {
    const int b = o >> 8, c = o & 255;
    const float* xr = sa3max + b * 512;
    const float* wr = w2 + (size_t)c * 512;
    float acc = 0.f;
    for (int k = 0; k < 512; ++k) acc += xr[k] * wr[k];
    const float s = g2[c] * rsqrtf(v2[c] + EPSBN);
    float y = (acc + b2[c] - m2[c]) * s + bt2[c];
    x2[o] = y > 0.f ? y : 0.f;
  }
  __syncthreads();
  for (int o = tid; o < 320; o += 512) {
    const int b = o / 40, c = o % 40;
    const float* xr = x2 + b * 256;
    const float* wr = w3 + (size_t)c * 256;
    float acc = b3[c];
    for (int k = 0; k < 256; ++k) acc += xr[k] * wr[k];
    out[o] = acc;
  }
}

// ---------------------------------------------------------------------------
// Host
// ---------------------------------------------------------------------------
static inline LayerP LP(void* const* d_in, int base) {
  LayerP p;
  p.w   = (const float*)d_in[base + 0];
  p.b   = (const float*)d_in[base + 1];
  p.g   = (const float*)d_in[base + 2];
  p.bta = (const float*)d_in[base + 3];
  p.m   = (const float*)d_in[base + 4];
  p.v   = (const float*)d_in[base + 5];
  return p;
}

extern "C" void kernel_launch(void* const* d_in, const int* in_sizes, int n_in,
                              void* d_out, int out_size, void* d_ws, size_t ws_size,
                              hipStream_t stream) {
  const float* xyz = (const float*)d_in[0];
  const float* fc2_w = (const float*)d_in[73];
  const float* fc2_b = (const float*)d_in[74];
  const float* bn2_g = (const float*)d_in[75];
  const float* bn2_b = (const float*)d_in[76];
  const float* bn2_m = (const float*)d_in[77];
  const float* bn2_v = (const float*)d_in[78];
  const float* fc3_w = (const float*)d_in[79];
  const float* fc3_b = (const float*)d_in[80];

  char* ws = (char*)d_ws;
  size_t off = 0;
  auto alloc = [&](size_t bytes) { char* p = ws + off; off = (off + bytes + 255) & ~(size_t)255; return p; };
  float* new_xyz = (float*)alloc((size_t)8 * 512 * 3 * 4);
  float* g1      = (float*)alloc((size_t)4096 * 16 * 3 * 4);
  float* g2      = (float*)alloc((size_t)4096 * 32 * 3 * 4);
  float* g3      = (float*)alloc((size_t)4096 * 128 * 3 * 4);
  float* feat    = (float*)alloc((size_t)4096 * 352 * 4);
  float* part    = (float*)alloc((size_t)128 * 512 * 4);
  float* sa3max  = (float*)alloc((size_t)8 * 512 * 4);
  (void)ws_size; (void)in_sizes; (void)n_in; (void)out_size;

  // 12 conv layers: {cout, cin, cinPad}; order: branch0 L0..2, branch1, branch2, sa3
  const int dims[12][3] = {
    {32, 3, 32},   {32, 32, 32},  {64, 32, 32},
    {64, 3, 32},   {64, 64, 64},  {128, 64, 64},
    {64, 3, 32},   {96, 64, 64},  {128, 96, 96},
    {128, 323, 352}, {256, 128, 128}, {512, 256, 256}
  };
  ConvL cl[12];
  for (int li = 0; li < 12; ++li) {
    const int cout = dims[li][0], cin = dims[li][1], cinPad = dims[li][2];
    _Float16* wh = (_Float16*)alloc((size_t)cout * cinPad * 2);
    float2*   st = (float2*)alloc((size_t)cout * sizeof(float2));
    const int base = (li < 9) ? (1 + li * 6) : (1 + 54 + (li - 9) * 6);
    LayerP p = LP(d_in, base);
    const int total = cout * cinPad;
    prep_layer_kernel<<<(total + 255) / 256, 256, 0, stream>>>(
        p.w, p.b, p.g, p.bta, p.m, p.v, wh, st, cout, cin, cinPad);
    cl[li].w = wh; cl[li].st = st;
  }

  fps_kernel<<<8, 512, 0, stream>>>(xyz, new_xyz, feat);
  ballquery_kernel<<<512, 256, 0, stream>>>(xyz, new_xyz, g1, g2, g3);
  sa1_branch_kernel<16, 32, 32, 64, 40, 40, 2><<<4096, 64, 0, stream>>>(g1, cl[0], cl[1], cl[2], feat, 3);
  sa1_branch_kernel<32, 64, 64, 128, 72, 72, 4><<<4096, 128, 0, stream>>>(g2, cl[3], cl[4], cl[5], feat, 67);
  sa1_branch_kernel<128, 64, 96, 128, 104, 72, 8><<<4096, 256, 0, stream>>>(g3, cl[6], cl[7], cl[8], feat, 195);
  sa3_kernel<<<128, 256, 0, stream>>>(feat, cl[9], cl[10], cl[11], part);
  sa3_reduce_kernel<<<16, 256, 0, stream>>>(part, sa3max);
  fc_kernel<<<1, 512, 0, stream>>>(sa3max, fc2_w, fc2_b, bn2_g, bn2_b, bn2_m, bn2_v,
                                   fc3_w, fc3_b, (float*)d_out);
}
